// Mamba2Base_13219909337560
// MI455X (gfx1250) — matrix operand order repaired and checked by reading, not compiled, not run on hardware
//
#include <hip/hip_runtime.h>
#include <math.h>

// ---------------- problem constants ----------------
#define HID       2048
#define INTER     4096
#define NH        64            // heads H
#define PD        64            // head dim P
#define NS        128           // state dim N
#define KCONV     4
#define CHK       256           // chunk length
#define CONV_DIM  (INTER + 2*NS)            // 4352
#define PROJ_DIM  (INTER + CONV_DIM + NH)   // 8512
#define SEQ       2048
#define BATCH     2
#define TOK       (BATCH*SEQ)   // 4096
#define NC        (SEQ/CHK)     // 8
#define PDP       72            // padded row stride for LDS tiles (bank spread)

// ---------------- WMMA plumbing (gfx1250, wave32) ----------------
typedef __attribute__((ext_vector_type(16))) __bf16 v16bf;
typedef __attribute__((ext_vector_type(8)))  float  v8f;

union Frag {
  v16bf v;
  uint4 q[2];
  unsigned short s[16];
};

__device__ inline v8f zero8() { v8f z; for (int i = 0; i < 8; ++i) z[i] = 0.f; return z; }

__device__ inline unsigned short f2bf(float f) {
  unsigned int u = __float_as_uint(f);
  u = (u + 0x7FFFu + ((u >> 16) & 1u)) >> 16;   // RNE
  return (unsigned short)u;
}
__device__ inline float bf2f(unsigned short s) {
  return __uint_as_float(((unsigned int)s) << 16);
}

__device__ inline v8f wmma_bf16(const Frag& a, const Frag& b, v8f c) {
  return __builtin_amdgcn_wmma_f32_16x16x32_bf16(false, a.v, false, b.v, (short)0, c, false, false);
}

// A operand 16x32 (M x K), row-major source, ld in elements.
// Lane m = lane&15, hi = lane>>4: k = hi*16 + 0..15 contiguous.
__device__ inline Frag loadA_rm(const unsigned short* base, int ld, int lane) {
  Frag f;
  int m = lane & 15, hi = lane >> 4;
  const unsigned short* p = base + (size_t)m * ld + hi * 16;
  f.q[0] = *(const uint4*)(p);
  f.q[1] = *(const uint4*)(p + 8);
  return f;
}

// B operand 32x16 (K x N): source row-major over [col][k] (row = output column),
// lane col = lane&15, k = (lane>>4)*16 + 0..15 contiguous.
__device__ inline Frag loadB_rm(const unsigned short* base, int ld, int lane) {
  Frag f;
  int c = lane & 15, hi = lane >> 4;
  const unsigned short* p = base + (size_t)c * ld + hi * 16;
  f.q[0] = *(const uint4*)(p);
  f.q[1] = *(const uint4*)(p + 8);
  return f;
}

// B operand gathered from an LDS tile [l][p] with row stride PDP: element(k=l, col=p)
__device__ inline Frag loadB_lds_tile(const unsigned short* sT, int l0, int p0, int lane) {
  Frag f;
  int col = lane & 15, hi = lane >> 4;
  for (int j = 0; j < 16; ++j)
    f.s[j] = sT[(l0 + hi * 16 + j) * PDP + p0 + col];
  return f;
}

// ---------------- kernels ----------------
__global__ __launch_bounds__(256) void cvt_f32_bf16(const float* __restrict__ src,
                                                    unsigned short* __restrict__ dst, size_t n) {
  size_t i = (size_t)blockIdx.x * blockDim.x + threadIdx.x;
  size_t stride = (size_t)gridDim.x * blockDim.x;
  for (; i < n; i += stride) dst[i] = f2bf(src[i]);
}

// C[m,n] = sum_k A[m,k] * B[n,k]; A: MxK, B: NxK (both bf16 row-major), C f32.
// 256 thr (8 waves). Block tile 128(M) x 64(N); wave tile 16 x 64.
// LDS-staged, double-buffered over 32-wide K steps.
#define GM 128
#define GN 64
#define GK 32
__global__ __launch_bounds__(256) void gemm_bf16_nt(const unsigned short* __restrict__ A,
                                                    const unsigned short* __restrict__ B,
                                                    float* __restrict__ C,
                                                    int M, int N, int K) {
  __shared__ __align__(16) unsigned short sA[2][GM * GK];   // 8KB x2
  __shared__ __align__(16) unsigned short sB[2][GN * GK];   // 4KB x2
  const int tid = threadIdx.x, lane = tid & 31, wave = tid >> 5;
  const int rowBlk = blockIdx.x * GM;
  const int colBlk = blockIdx.y * GN;

  auto stage = [&](int buf, int k0) {
    {
      int row = tid >> 1, kh = (tid & 1) * 16;
      const unsigned short* g = A + (size_t)(rowBlk + row) * K + k0 + kh;
      __builtin_prefetch(g + 2 * GK, 0, 3);                 // WGP-scope, next-next tile
      *(uint4*)&sA[buf][row * GK + kh]     = *(const uint4*)g;
      *(uint4*)&sA[buf][row * GK + kh + 8] = *(const uint4*)(g + 8);
    }
    {
      int row = tid >> 2, kq = (tid & 3) * 8;
      const unsigned short* g = B + (size_t)(colBlk + row) * K + k0 + kq;
      __builtin_prefetch(g + 2 * GK, 0, 3);
      *(uint4*)&sB[buf][row * GK + kq] = *(const uint4*)g;
    }
  };

  v8f acc[4];
  for (int t = 0; t < 4; ++t) acc[t] = zero8();

  const int nsteps = K / GK;
  stage(0, 0);
  __syncthreads();
  for (int step = 0; step < nsteps; ++step) {
    int buf = step & 1;
    if (step + 1 < nsteps) stage(buf ^ 1, (step + 1) * GK);   // overlap with math below
    // issue A + all four B fragment loads into distinct registers first so the
    // scheduler can clause the ds_loads and use partial dscnt waits
    Frag a  = loadA_rm(&sA[buf][(wave * 16) * GK], GK, lane);
    Frag b0 = loadB_rm(&sB[buf][ 0 * GK], GK, lane);
    Frag b1 = loadB_rm(&sB[buf][16 * GK], GK, lane);
    Frag b2 = loadB_rm(&sB[buf][32 * GK], GK, lane);
    Frag b3 = loadB_rm(&sB[buf][48 * GK], GK, lane);
    acc[0] = wmma_bf16(a, b0, acc[0]);
    acc[1] = wmma_bf16(a, b1, acc[1]);
    acc[2] = wmma_bf16(a, b2, acc[2]);
    acc[3] = wmma_bf16(a, b3, acc[3]);
    __syncthreads();
  }

  for (int t = 0; t < 4; ++t)
    for (int i = 0; i < 8; ++i) {
      int m = rowBlk + wave * 16 + (lane >> 4) * 8 + i;
      int n = colBlk + t * 16 + (lane & 15);
      C[(size_t)m * N + n] = acc[t][i];
    }
}

// depthwise causal conv(K=4)+SiLU; emits hs (raw) and hs*dt (both bf16), B, C, dt.
__global__ __launch_bounds__(256) void conv_silu_kernel(const float* __restrict__ proj,
                                                        const float* __restrict__ conv_w,
                                                        const float* __restrict__ conv_b,
                                                        const float* __restrict__ dt_bias,
                                                        unsigned short* __restrict__ hs_bf,
                                                        unsigned short* __restrict__ hsdt_bf,
                                                        unsigned short* __restrict__ Bm_bf,
                                                        unsigned short* __restrict__ Cm_bf,
                                                        float* __restrict__ dtp) {
  int t = blockIdx.x;
  int s = t & (SEQ - 1);
  int tid = threadIdx.x;
  __shared__ float sdtv[NH];
  if (tid < NH) {
    float x = proj[(size_t)t * PROJ_DIM + (INTER + CONV_DIM) + tid] + dt_bias[tid];
    float sp = (x > 20.f) ? x : log1pf(__expf(x));
    sp = fminf(fmaxf(sp, 0.001f), 100.f);
    dtp[(size_t)t * NH + tid] = sp;
    sdtv[tid] = sp;
  }
  __syncthreads();
  for (int ch = tid; ch < CONV_DIM; ch += 256) {
    float acc = conv_b[ch];
    for (int k = 0; k < KCONV; ++k) {
      int sp = s - (KCONV - 1) + k;
      if (sp >= 0)
        acc += conv_w[ch * KCONV + k] *
               proj[(size_t)(t - (KCONV - 1) + k) * PROJ_DIM + INTER + ch];
    }
    float v = acc / (1.f + __expf(-acc));       // SiLU
    if (ch < INTER) {
      hs_bf[(size_t)t * INTER + ch]   = f2bf(v);
      hsdt_bf[(size_t)t * INTER + ch] = f2bf(v * sdtv[ch >> 6]);
    } else if (ch < INTER + NS) {
      Bm_bf[(size_t)t * NS + (ch - INTER)] = f2bf(v);
    } else {
      Cm_bf[(size_t)t * NS + (ch - INTER - NS)] = f2bf(v);
    }
  }
}

// Per (b*chunk, head): Y_diag and chunk states, all matmuls on WMMA.
__global__ __launch_bounds__(256) void chunk_kernel(const unsigned short* __restrict__ hsdt_bf,
                                                    const unsigned short* __restrict__ Bm_bf,
                                                    const unsigned short* __restrict__ Cm_bf,
                                                    const float* __restrict__ dtp,
                                                    const float* __restrict__ A_log,
                                                    float* __restrict__ Acum_g,
                                                    float* __restrict__ ybuf,
                                                    float* __restrict__ Sbuf) {
  const int bc = blockIdx.x;          // b*NC + c  (0..15)
  const int h  = blockIdx.y;          // 0..63
  const int tid = threadIdx.x;
  const int lane = tid & 31, wave = tid >> 5;
  const int tok0 = bc * CHK;          // == b*SEQ + c*CHK

  __shared__ float sAc[CHK];
  __shared__ float sdec[CHK];
  __shared__ __align__(16) unsigned short sM[8][16 * 32];       // 8KB wave scratch
  __shared__ __align__(16) unsigned short sHdt[CHK * PDP];      // 36KB hs*dt tile (padded)

  float Ah = -__expf(A_log[h]);
  sAc[tid] = Ah * dtp[(size_t)(tok0 + tid) * NH + h];
  // stage hs*dt chunk tile: row tid, 64 cols
  for (int j = 0; j < 8; ++j)
    *(uint4*)&sHdt[tid * PDP + j * 8] =
        *(const uint4*)(hsdt_bf + (size_t)(tok0 + tid) * INTER + h * PD + j * 8);
  __syncthreads();
  if (tid == 0) {
    float run = 0.f;
    for (int l = 0; l < CHK; ++l) { run += sAc[l]; sAc[l] = run; }
  }
  __syncthreads();
  {
    float last = sAc[CHK - 1];
    sdec[tid] = __expf(last - sAc[tid]);
    Acum_g[((size_t)(bc * NH + h)) * CHK + tid] = sAc[tid];
  }
  __syncthreads();

  // ---- Y_diag: each wave owns two 16-row z tiles ----
  for (int t = 0; t < 2; ++t) {
    int zbase = (wave * 2 + t) * 16;
    Frag aC[4];                                     // hoisted C fragments (z rows)
    for (int nk = 0; nk < 4; ++nk)
      aC[nk] = loadA_rm(Cm_bf + (size_t)(tok0 + zbase) * NS + nk * 32, NS, lane);
    v8f accY[4];
    for (int i = 0; i < 4; ++i) accY[i] = zero8();
    for (int lb = 0; lb <= zbase; lb += 32) {       // causal l-blocks
      for (int half = 0; half < 2; ++half) {
        int lt = lb + half * 16;
        // issue all four B fragments before the WMMA chain
        Frag bb0 = loadB_rm(Bm_bf + (size_t)(tok0 + lt) * NS +  0, NS, lane);
        Frag bb1 = loadB_rm(Bm_bf + (size_t)(tok0 + lt) * NS + 32, NS, lane);
        Frag bb2 = loadB_rm(Bm_bf + (size_t)(tok0 + lt) * NS + 64, NS, lane);
        Frag bb3 = loadB_rm(Bm_bf + (size_t)(tok0 + lt) * NS + 96, NS, lane);
        v8f g = zero8();
        g = wmma_bf16(aC[0], bb0, g);
        g = wmma_bf16(aC[1], bb1, g);
        g = wmma_bf16(aC[2], bb2, g);
        g = wmma_bf16(aC[3], bb3, g);
        int lloc = lt + (lane & 15);
        for (int i = 0; i < 8; ++i) {               // M = G * L (mask + exp decay)
          int zloc = zbase + (lane >> 4) * 8 + i;
          float v = 0.f;
          if (lloc <= zloc) v = g[i] * __expf(sAc[zloc] - sAc[lloc]);
          sM[wave][(zloc - zbase) * 32 + (lloc - lb)] = f2bf(v);
        }
      }
      Frag aM = loadA_rm(&sM[wave][0], 32, lane);   // restage M into A layout
      Frag h0 = loadB_lds_tile(sHdt, lb,  0, lane);
      Frag h1 = loadB_lds_tile(sHdt, lb, 16, lane);
      Frag h2 = loadB_lds_tile(sHdt, lb, 32, lane);
      Frag h3 = loadB_lds_tile(sHdt, lb, 48, lane);
      accY[0] = wmma_bf16(aM, h0, accY[0]);
      accY[1] = wmma_bf16(aM, h1, accY[1]);
      accY[2] = wmma_bf16(aM, h2, accY[2]);
      accY[3] = wmma_bf16(aM, h3, accY[3]);
    }
    for (int pt = 0; pt < 4; ++pt)
      for (int i = 0; i < 8; ++i) {
        int z = zbase + (lane >> 4) * 8 + i;
        ybuf[(size_t)(tok0 + z) * INTER + h * PD + pt * 16 + (lane & 15)] = accY[pt][i];
      }
  }

  // ---- chunk states S[n,p] = sum_l (B[l,n]*dec[l]) * (hs[l,p]*dt[l]) ----
  {
    int nb = wave * 16;
    v8f accS[4];
    for (int i = 0; i < 4; ++i) accS[i] = zero8();
    for (int lb = 0; lb < CHK; lb += 32) {
      Frag aB;
      {
        int m = lane & 15, hi = lane >> 4;
        for (int i = 0; i < 8; ++i) {
          int kb = hi * 16 + i * 2;
          for (int d2 = 0; d2 < 2; ++d2) {
            int l = lb + kb + d2;
            float v = bf2f(Bm_bf[(size_t)(tok0 + l) * NS + nb + m]) * sdec[l];
            aB.s[2 * i + d2] = f2bf(v);
          }
        }
      }
      Frag h0 = loadB_lds_tile(sHdt, lb,  0, lane);
      Frag h1 = loadB_lds_tile(sHdt, lb, 16, lane);
      Frag h2 = loadB_lds_tile(sHdt, lb, 32, lane);
      Frag h3 = loadB_lds_tile(sHdt, lb, 48, lane);
      accS[0] = wmma_bf16(aB, h0, accS[0]);
      accS[1] = wmma_bf16(aB, h1, accS[1]);
      accS[2] = wmma_bf16(aB, h2, accS[2]);
      accS[3] = wmma_bf16(aB, h3, accS[3]);
    }
    size_t sb = ((size_t)(bc * NH + h)) * (NS * PD);
    for (int pt = 0; pt < 4; ++pt)
      for (int i = 0; i < 8; ++i) {
        int n = nb + (lane >> 4) * 8 + i;
        Sbuf[sb + (size_t)n * PD + pt * 16 + (lane & 15)] = accS[pt][i];
      }
  }
}

// Inter-chunk carry: carry_{c} = carry_{c-1}*exp(Atot_{c-1}) + S_{c-1}
__global__ __launch_bounds__(256) void carry_kernel(const float* __restrict__ Sbuf,
                                                    const float* __restrict__ Acum_g,
                                                    float* __restrict__ CSbuf) {
  int bh = blockIdx.x;                // b*NH + h
  int b = bh / NH, h = bh % NH;
  int tid = threadIdx.x;
  float carry[32];
  for (int j = 0; j < 32; ++j) carry[j] = 0.f;
  for (int c = 0; c < NC; ++c) {
    int bc = b * NC + c;
    size_t base = ((size_t)(bc * NH + h)) * (NS * PD);
    for (int j = 0; j < 32; ++j) CSbuf[base + tid + j * 256] = carry[j];
    float dc = __expf(Acum_g[((size_t)(bc * NH + h)) * CHK + (CHK - 1)]);
    for (int j = 0; j < 32; ++j) carry[j] = carry[j] * dc + Sbuf[base + tid + j * 256];
  }
}

// Y_off = (C . states) * exp(Acum)  +  D-residual, accumulated into y.
__global__ __launch_bounds__(256) void yoff_kernel(const unsigned short* __restrict__ hs_bf,
                                                   const unsigned short* __restrict__ Cm_bf,
                                                   const float* __restrict__ CSbuf,
                                                   const float* __restrict__ Acum_g,
                                                   const float* __restrict__ Dv,
                                                   float* __restrict__ ybuf) {
  int bc = blockIdx.x, h = blockIdx.y;
  int tid = threadIdx.x, lane = tid & 31, wave = tid >> 5;
  int tok0 = bc * CHK;
  size_t csb = ((size_t)(bc * NH + h)) * (NS * PD);
  float Dh = Dv[h];

  __shared__ __align__(16) unsigned short sCS[NS * PDP];   // 18KB state tile (bf16, padded)
  __shared__ float sAc[CHK];
  {
    int n = tid >> 1, half = (tid & 1) * 32;
    for (int j = 0; j < 32; ++j)
      sCS[n * PDP + half + j] = f2bf(CSbuf[csb + (size_t)n * PD + half + j]);
    sAc[tid] = Acum_g[((size_t)(bc * NH + h)) * CHK + tid];
  }
  __syncthreads();

  for (int t = 0; t < 2; ++t) {
    int lbase = (wave * 2 + t) * 16;
    v8f acc[4];
    for (int i = 0; i < 4; ++i) acc[i] = zero8();
    for (int nk = 0; nk < NS; nk += 32) {
      Frag a  = loadA_rm(Cm_bf + (size_t)(tok0 + lbase) * NS + nk, NS, lane);
      Frag b0 = loadB_lds_tile(sCS, nk,  0, lane);
      Frag b1 = loadB_lds_tile(sCS, nk, 16, lane);
      Frag b2 = loadB_lds_tile(sCS, nk, 32, lane);
      Frag b3 = loadB_lds_tile(sCS, nk, 48, lane);
      acc[0] = wmma_bf16(a, b0, acc[0]);
      acc[1] = wmma_bf16(a, b1, acc[1]);
      acc[2] = wmma_bf16(a, b2, acc[2]);
      acc[3] = wmma_bf16(a, b3, acc[3]);
    }
    for (int pt = 0; pt < 4; ++pt)
      for (int i = 0; i < 8; ++i) {
        int l = lbase + (lane >> 4) * 8 + i;
        int p = pt * 16 + (lane & 15);
        float sdo = __expf(sAc[l]);
        size_t yi = (size_t)(tok0 + l) * INTER + h * PD + p;
        float hsv = bf2f(hs_bf[yi]);
        ybuf[yi] += acc[pt][i] * sdo + Dh * hsv;
      }
  }
}

// y * SiLU(gate), RMSNorm, emit bf16 for the output projection
__global__ __launch_bounds__(256) void gate_norm_kernel(const float* __restrict__ proj,
                                                        const float* __restrict__ ybuf,
                                                        const float* __restrict__ norm_w,
                                                        unsigned short* __restrict__ yn_bf) {
  int t = blockIdx.x, tid = threadIdx.x;
  __shared__ float red[256];
  float ss = 0.f;
  for (int j = tid; j < INTER; j += 256) {
    float g = proj[(size_t)t * PROJ_DIM + j];
    float yg = ybuf[(size_t)t * INTER + j] * (g / (1.f + __expf(-g)));
    ss += yg * yg;
  }
  red[tid] = ss;
  __syncthreads();
  for (int st = 128; st > 0; st >>= 1) {
    if (tid < st) red[tid] += red[tid + st];
    __syncthreads();
  }
  float scale = rsqrtf(red[0] / (float)INTER + 1e-5f);
  for (int j = tid; j < INTER; j += 256) {
    float g = proj[(size_t)t * PROJ_DIM + j];
    float yg = ybuf[(size_t)t * INTER + j] * (g / (1.f + __expf(-g)));
    yn_bf[(size_t)t * INTER + j] = f2bf(yg * scale * norm_w[j]);
  }
}

// ---------------- launcher ----------------
extern "C" void kernel_launch(void* const* d_in, const int* in_sizes, int n_in,
                              void* d_out, int out_size, void* d_ws, size_t ws_size,
                              hipStream_t stream) {
  (void)in_sizes; (void)n_in; (void)out_size; (void)ws_size;
  const float* x     = (const float*)d_in[0];
  const float* Wi    = (const float*)d_in[1];
  const float* convw = (const float*)d_in[2];
  const float* convb = (const float*)d_in[3];
  const float* dtb   = (const float*)d_in[4];
  const float* Alog  = (const float*)d_in[5];
  const float* Dv    = (const float*)d_in[6];
  const float* nw    = (const float*)d_in[7];
  const float* Wo    = (const float*)d_in[8];
  float* out = (float*)d_out;

  char* w = (char*)d_ws;
  size_t off = 0;
  auto alloc = [&](size_t bytes) -> char* {
    char* p = w + off;
    off = (off + bytes + 255) & ~(size_t)255;
    return p;
  };
  unsigned short* Xbf   = (unsigned short*)alloc((size_t)TOK * HID * 2);
  unsigned short* Wibf  = (unsigned short*)alloc((size_t)PROJ_DIM * HID * 2);
  unsigned short* Wobf  = (unsigned short*)alloc((size_t)HID * INTER * 2);
  float*          proj  = (float*)         alloc((size_t)TOK * PROJ_DIM * 4);
  unsigned short* hsbf  = (unsigned short*)alloc((size_t)TOK * INTER * 2);
  unsigned short* hsdt  = (unsigned short*)alloc((size_t)TOK * INTER * 2);
  unsigned short* Bmbf  = (unsigned short*)alloc((size_t)TOK * NS * 2);
  unsigned short* Cmbf  = (unsigned short*)alloc((size_t)TOK * NS * 2);
  float*          dtp   = (float*)         alloc((size_t)TOK * NH * 4);
  float*          Acum  = (float*)         alloc((size_t)BATCH * NC * NH * CHK * 4);
  float*          ybuf  = (float*)         alloc((size_t)TOK * INTER * 4);
  float*          Sbuf  = (float*)         alloc((size_t)BATCH * NC * NH * NS * PD * 4);
  float*          CSbf  = (float*)         alloc((size_t)BATCH * NC * NH * NS * PD * 4);
  unsigned short* ynbf  = (unsigned short*)alloc((size_t)TOK * INTER * 2);

  cvt_f32_bf16<<<1024, 256, 0, stream>>>(x,  Xbf,  (size_t)TOK * HID);
  cvt_f32_bf16<<<2048, 256, 0, stream>>>(Wi, Wibf, (size_t)PROJ_DIM * HID);
  cvt_f32_bf16<<<1024, 256, 0, stream>>>(Wo, Wobf, (size_t)HID * INTER);

  gemm_bf16_nt<<<dim3(TOK / 128, PROJ_DIM / 64), 256, 0, stream>>>(
      Xbf, Wibf, proj, TOK, PROJ_DIM, HID);

  conv_silu_kernel<<<TOK, 256, 0, stream>>>(proj, convw, convb, dtb,
                                            hsbf, hsdt, Bmbf, Cmbf, dtp);

  chunk_kernel<<<dim3(BATCH * NC, NH), 256, 0, stream>>>(
      hsdt, Bmbf, Cmbf, dtp, Alog, Acum, ybuf, Sbuf);

  carry_kernel<<<BATCH * NH, 256, 0, stream>>>(Sbuf, Acum, CSbf);

  yoff_kernel<<<dim3(BATCH * NC, NH), 256, 0, stream>>>(hsbf, Cmbf, CSbf, Acum, Dv, ybuf);

  gate_norm_kernel<<<TOK, 256, 0, stream>>>(proj, ybuf, nw, ynbf);

  gemm_bf16_nt<<<dim3(TOK / 128, HID / 64), 256, 0, stream>>>(
      ynbf, Wobf, out, TOK, HID, INTER);
}
